// SIREConvBase_2645699854684
// MI455X (gfx1250) — compile-verified, matrix operand order checked by reading, not run on hardware
//
#include <hip/hip_runtime.h>
#include <hip/hip_bf16.h>

#define DD 128
#define LDSS_H 136  // padded LDS row stride in ushorts (272B: bank-safe, 16B aligned)

typedef __attribute__((ext_vector_type(16))) __bf16 v16bf;
typedef __attribute__((ext_vector_type(16))) unsigned short v16us;
typedef __attribute__((ext_vector_type(8))) unsigned short v8us;
typedef __attribute__((ext_vector_type(8))) float v8f;

__device__ __forceinline__ v8f wmma_bf16(v16us a, v16us b, v8f c) {
  return __builtin_amdgcn_wmma_f32_16x16x32_bf16(
      false, __builtin_bit_cast(v16bf, a),
      false, __builtin_bit_cast(v16bf, b),
      (short)0, c, false, false);
}

// Assemble a 16-element bf16 fragment from LDS: elements 0..7 at p[0..7],
// elements 8..15 at p[16..23] (matches A/B 16-bit WMMA VGPR layout).
__device__ __forceinline__ v16us load_frag(const unsigned short* p) {
  v8us a = *(const v8us*)(p);
  v8us b = *(const v8us*)(p + 16);
  return __builtin_shufflevector(a, b, 0, 1, 2, 3, 4, 5, 6, 7, 8, 9, 10, 11,
                                 12, 13, 14, 15);
}

// Truncation-based hi/lo bf16 split of 8 floats -> packed LDS stores.
// lo = f - hi is exact; truncating lo leaves ~2^-17 relative error overall.
__device__ __forceinline__ void stage8(const float* gsrc, unsigned short* ldsH,
                                       unsigned short* ldsL, int ofs) {
  const float4* gp = (const float4*)gsrc;
  float4 a0 = gp[0], a1 = gp[1];
  float f[8] = {a0.x, a0.y, a0.z, a0.w, a1.x, a1.y, a1.z, a1.w};
  unsigned int hp[4], lp[4];
#pragma unroll
  for (int i = 0; i < 4; ++i) {
    float f0 = f[2 * i], f1 = f[2 * i + 1];
    unsigned int u0 = __float_as_uint(f0);
    unsigned int u1 = __float_as_uint(f1);
    hp[i] = (u0 >> 16) | (u1 & 0xFFFF0000u);
    float l0 = f0 - __uint_as_float(u0 & 0xFFFF0000u);
    float l1 = f1 - __uint_as_float(u1 & 0xFFFF0000u);
    lp[i] = (__float_as_uint(l0) >> 16) | (__float_as_uint(l1) & 0xFFFF0000u);
  }
  *(uint4*)(ldsH + ofs) = make_uint4(hp[0], hp[1], hp[2], hp[3]);
  *(uint4*)(ldsL + ofs) = make_uint4(lp[0], lp[1], lp[2], lp[3]);
}

// Per-element truncation split for the register-resident B (one-time cost).
__device__ __forceinline__ void split_bf16(float f, unsigned short& hi,
                                           unsigned short& lo) {
  unsigned int u = __float_as_uint(f);
  hi = (unsigned short)(u >> 16);
  float l = f - __uint_as_float(u & 0xFFFF0000u);
  lo = (unsigned short)(__float_as_uint(l) >> 16);
}

// ------------------------------ setup kernels ------------------------------
__global__ void zero_kernel(float* __restrict__ out, float* __restrict__ norm,
                            int N) {
  long i = (long)blockIdx.x * blockDim.x + threadIdx.x;
  long total = (long)N * DD;
  if (i < total) out[i] = 0.0f;
  if (i < N) norm[i] = 0.0f;
}

__global__ void deg_kernel(const int* __restrict__ dst,
                           float* __restrict__ norm, int E) {
  int i = blockIdx.x * blockDim.x + threadIdx.x;
  if (i < E) atomicAdd(norm + dst[i], 1.0f);
}

__global__ void norm_kernel(float* __restrict__ norm, int N) {
  int i = blockIdx.x * blockDim.x + threadIdx.x;
  if (i < N) norm[i] = rsqrtf(fmaxf(norm[i], 1.0f));
}

// -------------------- node projections: q = nfeat@Wq + b, k = nfeat@Wk -----
__global__ __launch_bounds__(256) void node_kernel(
    const float* __restrict__ nfeat, const float* __restrict__ Wq,
    const float* __restrict__ Wk, const float* __restrict__ bias,
    float* __restrict__ qbuf, float* __restrict__ kbuf, int N) {
  __shared__ __align__(16) unsigned short ldsH[16 * LDSS_H];
  __shared__ __align__(16) unsigned short ldsL[16 * LDSS_H];

  const int tid = threadIdx.x;
  const int lane = tid & 31;
  const int w = tid >> 5;
  const int nlo = lane & 15;
  const int khalf = lane >> 4;
  const int ncol = w * 16 + nlo;

  const bool is_q = (blockIdx.y == 0);
  const float* __restrict__ W = is_q ? Wq : Wk;
  float* __restrict__ outp = is_q ? qbuf : kbuf;

  // B = W[K][ncol] fragments in registers (hi/lo bf16 split, one-time).
  v16us Bhi[4], Blo[4];
#pragma unroll
  for (int c = 0; c < 4; ++c) {
    const int kbase = khalf * 8 + 32 * c;
#pragma unroll
    for (int e = 0; e < 16; ++e) {
      int kk = kbase + e + (e & 8);
      unsigned short h, l;
      split_bf16(W[kk * DD + ncol], h, l);
      Bhi[c][e] = h;
      Blo[c][e] = l;
    }
  }

  const int tile = blockIdx.x;
  {
    int m = tid >> 4;
    int c8 = (tid & 15) << 3;
    int row = tile * 16 + m;
    if (row >= N) row = N - 1;
    stage8(nfeat + (long)row * DD + c8, ldsH, ldsL, m * LDSS_H + c8);
  }
  __syncthreads();

  v8f acc = {};
#pragma unroll
  for (int c = 0; c < 4; ++c) {
    const int kbase = khalf * 8 + 32 * c;
    const int fofs = nlo * LDSS_H + kbase;
    v16us Ahi = load_frag(ldsH + fofs);
    v16us Alo = load_frag(ldsL + fofs);
    acc = wmma_bf16(Ahi, Bhi[c], acc);
    acc = wmma_bf16(Ahi, Blo[c], acc);
    acc = wmma_bf16(Alo, Bhi[c], acc);
  }

  const float bb = is_q ? bias[ncol] : 0.0f;
#pragma unroll
  for (int v = 0; v < 8; ++v) {
    int m = v + khalf * 8;
    int row = tile * 16 + m;
    if (row < N) outp[(long)row * DD + ncol] = acc[v] + bb;
  }
}

// -------- fused edge kernel: relu(q[dst]+k[src]+efeat@We)*norms -> scatter --
__global__ __launch_bounds__(256) void edge_kernel(
    const float* __restrict__ efeat, const int* __restrict__ src,
    const int* __restrict__ dst, const float* __restrict__ We,
    const float* __restrict__ qbuf, const float* __restrict__ kbuf,
    const float* __restrict__ norm, float* __restrict__ out, int E) {
  __shared__ __align__(16) unsigned short ldsH[16 * LDSS_H];
  __shared__ __align__(16) unsigned short ldsL[16 * LDSS_H];

  const int tid = threadIdx.x;
  const int lane = tid & 31;
  const int w = tid >> 5;
  const int nlo = lane & 15;
  const int khalf = lane >> 4;
  const int ncol = w * 16 + nlo;

  // We fragments resident in registers for the whole block (served from L2).
  v16us Bhi[4], Blo[4];
#pragma unroll
  for (int c = 0; c < 4; ++c) {
    const int kbase = khalf * 8 + 32 * c;
#pragma unroll
    for (int e = 0; e < 16; ++e) {
      int kk = kbase + e + (e & 8);
      unsigned short h, l;
      split_bf16(We[kk * DD + ncol], h, l);
      Bhi[c][e] = h;
      Blo[c][e] = l;
    }
  }

  const int m_stage = tid >> 4;
  const int c8_stage = (tid & 15) << 3;
  const int ntiles = (E + 15) >> 4;

  for (int tile = blockIdx.x; tile < ntiles; tile += gridDim.x) {
    // Stage + convert efeat tile (16 edges x 128) to bf16 hi/lo in LDS.
    {
      int erow = tile * 16 + m_stage;
      if (erow >= E) erow = E - 1;
      stage8(efeat + (long)erow * DD + c8_stage, ldsH, ldsL,
             m_stage * LDSS_H + c8_stage);
      // Prefetch next tile's rows (global_prefetch path, hides HBM latency).
      int ntile = tile + gridDim.x;
      if (ntile < ntiles) {
        int prow = ntile * 16 + m_stage;
        if (prow >= E) prow = E - 1;
        __builtin_prefetch(efeat + (long)prow * DD + c8_stage, 0, 0);
      }
    }
    __syncthreads();

    v8f acc = {};
#pragma unroll
    for (int c = 0; c < 4; ++c) {
      const int kbase = khalf * 8 + 32 * c;
      const int fofs = nlo * LDSS_H + kbase;
      v16us Ahi = load_frag(ldsH + fofs);
      v16us Alo = load_frag(ldsL + fofs);
      acc = wmma_bf16(Ahi, Bhi[c], acc);
      acc = wmma_bf16(Ahi, Blo[c], acc);
      acc = wmma_bf16(Alo, Bhi[c], acc);
    }

    // Fused epilogue: +q[dst]+k[src] (b folded into q), relu, norm, scatter.
#pragma unroll
    for (int v = 0; v < 8; ++v) {
      int m = v + khalf * 8;
      int e_idx = tile * 16 + m;
      if (e_idx < E) {
        int dd = dst[e_idx];
        int ss = src[e_idx];
        float val = acc[v] + qbuf[(long)dd * DD + ncol] +
                    kbuf[(long)ss * DD + ncol];
        val = fmaxf(val, 0.0f) * (norm[ss] * norm[dd]);
        atomicAdd(out + (long)dd * DD + ncol, val);
      }
    }
    __syncthreads();
  }
}

// ------------------------------ launcher -----------------------------------
extern "C" void kernel_launch(void* const* d_in, const int* in_sizes, int n_in,
                              void* d_out, int out_size, void* d_ws,
                              size_t ws_size, hipStream_t stream) {
  const float* nfeat = (const float*)d_in[0];
  const float* efeat = (const float*)d_in[1];
  const int* src = (const int*)d_in[2];
  const int* dst = (const int*)d_in[3];
  const float* Wq = (const float*)d_in[4];
  const float* Wk = (const float*)d_in[5];
  const float* We = (const float*)d_in[6];
  const float* bias = (const float*)d_in[7];
  float* out = (float*)d_out;

  const int N = in_sizes[0] / DD;
  const int E = in_sizes[2];

  float* ws = (float*)d_ws;
  float* qbuf = ws;                     // N*128 floats
  float* kbuf = qbuf + (size_t)N * DD;  // N*128 floats
  float* norm = kbuf + (size_t)N * DD;  // N floats (deg, then norm in place)

  dim3 blk(256);

  long zt = (long)N * DD;
  zero_kernel<<<dim3((unsigned)((zt + 255) / 256)), blk, 0, stream>>>(out, norm,
                                                                      N);
  deg_kernel<<<dim3((E + 255) / 256), blk, 0, stream>>>(dst, norm, E);
  norm_kernel<<<dim3((N + 255) / 256), blk, 0, stream>>>(norm, N);

  node_kernel<<<dim3((N + 15) / 16, 2), blk, 0, stream>>>(nfeat, Wq, Wk, bias,
                                                          qbuf, kbuf, N);

  int ntiles = (E + 15) >> 4;
  int nblk = ntiles < 1280 ? ntiles : 1280;  // persistent blocks
  edge_kernel<<<dim3(nblk), blk, 0, stream>>>(efeat, src, dst, We, qbuf, kbuf,
                                              norm, out, E);
}
